// LNN_11819749999118
// MI455X (gfx1250) — matrix operand order repaired and checked by reading, not compiled, not run on hardware
//
#include <hip/hip_runtime.h>
#include <math.h>

// Problem constants
#define NH      256
#define NS      16                 // samples per tile (one WMMA N-tile)
#define BTOT    131072
#define NTILES  (BTOT / NS)        // 8192
#define NWG     1024
#define TPW     (NTILES / NWG)     // 8 tiles per workgroup
#define THREADS 256                // 8 wave32

typedef __attribute__((ext_vector_type(16))) __bf16      v16bf;
typedef __attribute__((ext_vector_type(8)))  float       v8f;
typedef __attribute__((ext_vector_type(8)))  unsigned int v8u;

// ---- LDS layout (dword offsets) ----
#define OFF_W1   0                       // 1024 f32 : W1[i*4+c]
#define OFF_B1   1024                    // 256 f32
#define OFF_B2   1280                    // 256 f32
#define OFF_W3   1536                    // 256 f32
#define OFF_X    1792                    // 64 f32 : x[s*4+c] (16B aligned)
#define OFF_SIG  1856                    // 4096 f32 : sigma1[i*16+s]
#define OFF_ACC  5952                    // 192 f32 : acc[s*12 + slot] (0..3 J, 4..10 H pairs)
#define OFF_W2   6144                    // 256*129 dwords: bf16 pairs along i, row stride 129
#define OFF_XP   (OFF_W2 + 256*129)      // 5 groups * 128*17 dwords: bf16 pairs along k
#define OFF_UP   (OFF_XP + 5*128*17)     // 128*17 dwords: u packed bf16 pairs along j
#define LDS_DW   (OFF_UP + 128*17)       // 52224 dwords
#define LDS_BYTES (LDS_DW * 4)           // 208896 bytes (< 320 KB/WGP)

__device__ __forceinline__ unsigned int f2bf(float f) {
  unsigned int u = __float_as_uint(f);
  return (u + 0x7FFFu + ((u >> 16) & 1u)) >> 16;   // RNE fp32 -> bf16
}
__device__ __forceinline__ unsigned int packbf(float lo, float hi) {
  return f2bf(lo) | (f2bf(hi) << 16);
}
__device__ __forceinline__ float sigm(float z) { return 1.0f / (1.0f + __expf(-z)); }
__device__ __forceinline__ float sftp(float z) { return (z > 15.0f) ? z : log1pf(__expf(z)); }

__device__ __forceinline__ v8f wmma_bf16(v8u a, v8u b, v8f c) {
  return __builtin_amdgcn_wmma_f32_16x16x32_bf16(
      false, __builtin_bit_cast(v16bf, a),
      false, __builtin_bit_cast(v16bf, b),
      (short)0, c, false, false);
}

extern "C" __global__ __launch_bounds__(THREADS, 1)
void lnn_hess_solve(const float* __restrict__ x,
                    const float* __restrict__ W1g, const float* __restrict__ b1g,
                    const float* __restrict__ W2g, const float* __restrict__ b2g,
                    const float* __restrict__ W3g,
                    float* __restrict__ out)
{
  extern __shared__ unsigned int lds[];
  float* fl = (float*)lds;

  const int tid  = threadIdx.x;
  const int lane = tid & 31;
  const int wave = tid >> 5;
  const int lh   = lane >> 4;     // which 16-lane half
  const int ln   = lane & 15;

  // ---------- one-time per WG: stage weights; W2 -> packed bf16 in LDS ----------
  for (int idx = tid; idx < 1024; idx += THREADS) fl[OFF_W1 + idx] = W1g[idx];
  for (int idx = tid; idx < 256;  idx += THREADS) {
    fl[OFF_B1 + idx] = b1g[idx];
    fl[OFF_B2 + idx] = b2g[idx];
    fl[OFF_W3 + idx] = W3g[idx];
  }
  for (int idx = tid; idx < 256 * 128; idx += THREADS) {
    if (idx + THREADS < 256 * 128) __builtin_prefetch(W2g + 2 * (idx + THREADS), 0, 1);
    const int j = idx >> 7, i2 = idx & 127;
    const float2 w = ((const float2*)W2g)[idx];          // W2[j, 2*i2], W2[j, 2*i2+1]
    lds[OFF_W2 + j * 129 + i2] = packbf(w.x, w.y);
  }
  __syncthreads();

  static const int pc_[7] = {2, 2, 2, 2, 3, 3, 3};
  static const int pd_[7] = {0, 1, 2, 3, 0, 1, 3};

  for (int t = 0; t < TPW; ++t) {
    const int tile = blockIdx.x * TPW + t;
    const int s0   = tile * NS;                          // first sample of tile

    __syncthreads();                                     // protect reuse from prev tile
    if (tid < 192) fl[OFF_ACC + tid] = 0.0f;
    if (tid < NS)  ((float4*)(fl + OFF_X))[tid] = ((const float4*)x)[s0 + tid];
    __syncthreads();

    // ---------- build z1 / sigma1 / X = [softplus(z1) | sigma1*W1[:,c]] (bf16) ----------
    {
      const int s = tid & 15;
      const int ibase = (tid >> 4) * 16;
      const float4 xs = ((const float4*)(fl + OFF_X))[s];
      #pragma unroll
      for (int io = 0; io < 16; io += 2) {
        float sg[2], hp[2];
        float4 wv[2];
        #pragma unroll
        for (int k = 0; k < 2; ++k) {
          const int i = ibase + io + k;
          wv[k] = ((const float4*)(fl + OFF_W1))[i];
          const float z = wv[k].x * xs.x + wv[k].y * xs.y + wv[k].z * xs.z + wv[k].w * xs.w
                        + fl[OFF_B1 + i];
          sg[k] = sigm(z);
          hp[k] = sftp(z);
          fl[OFF_SIG + i * 16 + s] = sg[k];
        }
        const int k2 = (ibase + io) >> 1;
        lds[OFF_XP + 0 * 2176 + k2 * 17 + s] = packbf(hp[0],          hp[1]);
        lds[OFF_XP + 1 * 2176 + k2 * 17 + s] = packbf(sg[0] * wv[0].x, sg[1] * wv[1].x);
        lds[OFF_XP + 2 * 2176 + k2 * 17 + s] = packbf(sg[0] * wv[0].y, sg[1] * wv[1].y);
        lds[OFF_XP + 3 * 2176 + k2 * 17 + s] = packbf(sg[0] * wv[0].z, sg[1] * wv[1].z);
        lds[OFF_XP + 4 * 2176 + k2 * 17 + s] = packbf(sg[0] * wv[0].w, sg[1] * wv[1].w);
      }
    }
    __syncthreads();

    // ---------- Phase A: [Z2 | Q0..Q3] = W2 @ X  (wave owns 2 row-tiles x 5 groups) ----------
    #pragma unroll
    for (int rti = 0; rti < 2; ++rti) {
      const int rt   = wave * 2 + rti;
      const int jrow = rt * 16 + ln;                     // A-fragment row (M = lane&15)
      unsigned int Afr[8][8];                            // A frags for full K, reused over 5 groups
      #pragma unroll
      for (int kk = 0; kk < 8; ++kk)
        #pragma unroll
        for (int v = 0; v < 8; ++v)
          Afr[kk][v] = lds[OFF_W2 + jrow * 129 + kk * 16 + (v >> 2) * 8 + lh * 4 + (v & 3)];

      float q[4][8];
      float bb[8];
      #pragma unroll
      for (int g = 0; g < 5; ++g) {
        v8f c = {0.f, 0.f, 0.f, 0.f, 0.f, 0.f, 0.f, 0.f};
        #pragma unroll
        for (int kk = 0; kk < 8; ++kk) {
          v8u au, bu;
          #pragma unroll
          for (int v = 0; v < 8; ++v) {
            au[v] = Afr[kk][v];
            bu[v] = lds[OFF_XP + g * 2176 + (kk * 16 + (v >> 2) * 8 + lh * 4 + (v & 3)) * 17 + ln];
          }
          c = wmma_bf16(au, bu, c);
        }
        if (g == 0) {                                    // Z2 tile -> u (LDS bf16), b (regs)
          float uu[8];
          #pragma unroll
          for (int r = 0; r < 8; ++r) {
            const int j = rt * 16 + lh * 8 + r;          // C layout: M = r + 8*lh
            const float z2  = c[r] + fl[OFF_B2 + j];
            const float s2  = sigm(z2);
            const float w3j = fl[OFF_W3 + j];
            uu[r] = w3j * s2;
            bb[r] = w3j * s2 * (1.0f - s2);
          }
          #pragma unroll
          for (int pi = 0; pi < 4; ++pi)
            lds[OFF_UP + (rt * 8 + lh * 4 + pi) * 17 + ln] = packbf(uu[2 * pi], uu[2 * pi + 1]);
        } else {
          #pragma unroll
          for (int r = 0; r < 8; ++r) q[g - 1][r] = c[r];
        }
      }
      // Hb[c,d] += sum_j b * Q[:,c] * Q[:,d]  (rows 2,3 of H; all in-register per wave)
      #pragma unroll
      for (int p = 0; p < 7; ++p) {
        float acc = 0.0f;
        #pragma unroll
        for (int r = 0; r < 8; ++r) acc += bb[r] * q[pc_[p]][r] * q[pd_[p]][r];
        acc += __shfl_xor(acc, 16);
        if (lh == 0) atomicAdd(&fl[OFF_ACC + ln * 12 + 4 + p], acc);
      }
    }
    __syncthreads();

    // ---------- Phase B: V = W2^T @ U ; accumulate J and W1^T diag(a) W1 ----------
    {
      const unsigned short* W2h = (const unsigned short*)(lds + OFF_W2);  // row stride 258 halves
      #pragma unroll
      for (int rti = 0; rti < 2; ++rti) {
        const int rt   = wave * 2 + rti;
        const int irow = rt * 16 + ln;                   // column of W2 = row of V
        v8f c = {0.f, 0.f, 0.f, 0.f, 0.f, 0.f, 0.f, 0.f};
        #pragma unroll
        for (int kk = 0; kk < 8; ++kk) {
          v8u au, bu;
          #pragma unroll
          for (int v = 0; v < 8; ++v) {
            const int j = kk * 32 + (v >> 2) * 16 + lh * 8 + (v & 3) * 2;  // K index (even)
            const unsigned int lo = W2h[(j    ) * 258 + irow];             // W2[j,  i]
            const unsigned int hi = W2h[(j + 1) * 258 + irow];             // W2[j+1,i]
            au[v] = lo | (hi << 16);
            bu[v] = lds[OFF_UP + (kk * 16 + (v >> 2) * 8 + lh * 4 + (v & 3)) * 17 + ln];
          }
          c = wmma_bf16(au, bu, c);
        }
        float jac[4] = {0.f, 0.f, 0.f, 0.f};
        float ha[7]  = {0.f, 0.f, 0.f, 0.f, 0.f, 0.f, 0.f};
        #pragma unroll
        for (int r = 0; r < 8; ++r) {
          const int i    = rt * 16 + lh * 8 + r;
          const float vv = c[r];
          const float sg = fl[OFF_SIG + i * 16 + ln];
          const float rr = vv * sg;                       // r = v .* sigma1
          const float a  = rr * (1.0f - sg);              // a = v .* sigma1'
          const float4 w = ((const float4*)(fl + OFF_W1))[i];
          const float wc[4] = {w.x, w.y, w.z, w.w};
          #pragma unroll
          for (int cI = 0; cI < 4; ++cI) jac[cI] += wc[cI] * rr;
          #pragma unroll
          for (int p = 0; p < 7; ++p) ha[p] += a * wc[pc_[p]] * wc[pd_[p]];
        }
        #pragma unroll
        for (int cI = 0; cI < 4; ++cI) {
          float acc = jac[cI]; acc += __shfl_xor(acc, 16);
          if (lh == 0) atomicAdd(&fl[OFF_ACC + ln * 12 + cI], acc);
        }
        #pragma unroll
        for (int p = 0; p < 7; ++p) {
          float acc = ha[p]; acc += __shfl_xor(acc, 16);
          if (lh == 0) atomicAdd(&fl[OFF_ACC + ln * 12 + 4 + p], acc);
        }
      }
    }
    __syncthreads();

    // ---------- 2x2 pinv solve and store [qt, qtt] ----------
    if (tid < NS) {
      const int s = tid;
      const float* A = fl + OFF_ACC + s * 12;
      const float J0 = A[0], J1 = A[1];
      const float H20 = A[4], H21 = A[5], H22 = A[6], H23 = A[7];
      const float H30 = A[8], H31 = A[9], H33 = A[10];
      const float qt0 = fl[OFF_X + s * 4 + 2];
      const float qt1 = fl[OFF_X + s * 4 + 3];
      const float r0 = J0 - (H20 * qt0 + H21 * qt1);
      const float r1 = J1 - (H30 * qt0 + H31 * qt1);
      const float det = H22 * H33 - H23 * H23;
      const float fro = H22 * H22 + 2.0f * H23 * H23 + H33 * H33;
      float q0, q1;
      if (fabsf(det) > 1e-10f * fro) {                    // well-conditioned: exact inverse
        const float inv = 1.0f / det;
        q0 = ( H33 * r0 - H23 * r1) * inv;
        q1 = (-H23 * r0 + H22 * r1) * inv;
      } else if (fro > 0.0f) {                            // rank-1 symmetric pinv = B / ||B||_F^2
        q0 = (H22 * r0 + H23 * r1) / fro;
        q1 = (H23 * r0 + H33 * r1) / fro;
      } else { q0 = 0.0f; q1 = 0.0f; }
      float4 o; o.x = qt0; o.y = qt1; o.z = q0; o.w = q1;
      ((float4*)out)[s0 + s] = o;
    }
  }
}

extern "C" void kernel_launch(void* const* d_in, const int* in_sizes, int n_in,
                              void* d_out, int out_size, void* d_ws, size_t ws_size,
                              hipStream_t stream) {
  const float* x  = (const float*)d_in[0];
  const float* W1 = (const float*)d_in[1];
  const float* b1 = (const float*)d_in[2];
  const float* W2 = (const float*)d_in[3];
  const float* b2 = (const float*)d_in[4];
  const float* W3 = (const float*)d_in[5];
  // d_in[6] = b3 : additive constant, vanishes in all derivatives -> unused
  lnn_hess_solve<<<NWG, THREADS, LDS_BYTES, stream>>>(x, W1, b1, W2, b2, W3, (float*)d_out);
}
